// MoEFFN_49460843381514
// MI455X (gfx1250) — matrix operand order repaired and checked by reading, not compiled, not run on hardware
//
#include <hip/hip_runtime.h>

// ---------------- types ----------------
typedef __attribute__((ext_vector_type(16))) __bf16 v16bf;
typedef __attribute__((ext_vector_type(8)))  float  v8f;
typedef __attribute__((ext_vector_type(4)))  unsigned int v4u;

union Frag { v4u u[2]; v16bf v; };

#define N_TOK   16384
#define C_DIM   512
#define H_DIM   2048
#define E_NUM   8
#define TILE_M  32
#define MAX_TILES 1032          // 32768/32 + 8

__device__ __forceinline__ unsigned short f2bf(float f) {
    return __builtin_bit_cast(unsigned short, (__bf16)f);
}

__device__ __forceinline__ v8f wmma_bf16(const Frag& a, const Frag& b, v8f c) {
    return __builtin_amdgcn_wmma_f32_16x16x32_bf16(
        false, a.v, false, b.v, (short)0, c, false, false);
}

// ---------------- 1. convert x to bf16 (vectorized x4) ----------------
__global__ __launch_bounds__(256) void cvt_x_kernel(const float* __restrict__ x,
                                                    unsigned short* __restrict__ xb) {
    int i = blockIdx.x * 256 + threadIdx.x;           // over N*C/4 groups
    const float4 f = ((const float4*)x)[i];
    ushort4 o;
    o.x = f2bf(f.x); o.y = f2bf(f.y); o.z = f2bf(f.z); o.w = f2bf(f.w);
    ((ushort4*)xb)[i] = o;
}

// ---------------- 2. batched transpose + convert: [b][R][C] f32 -> [b][C][R] bf16 ----
__global__ __launch_bounds__(256) void transpose_bf16_kernel(const float* __restrict__ src,
                                                             unsigned short* __restrict__ dst,
                                                             int R, int C) {
    __shared__ unsigned short tile[32][33];
    const int b = blockIdx.z;
    const float* s = src + (size_t)b * R * C;
    unsigned short* d = dst + (size_t)b * R * C;
    const int ct = blockIdx.x * 32, rt = blockIdx.y * 32;
    const int tx = threadIdx.x & 31, ty = threadIdx.x >> 5;   // 32x8
    #pragma unroll
    for (int j = 0; j < 32; j += 8)
        tile[ty + j][tx] = f2bf(s[(size_t)(rt + ty + j) * C + ct + tx]);
    __syncthreads();
    #pragma unroll
    for (int j = 0; j < 32; j += 8)
        d[(size_t)(ct + ty + j) * R + rt + tx] = tile[tx][ty + j];
}

// ---------------- 3. router: one wave32 per token ----------------
__global__ __launch_bounds__(256) void router_kernel(const float* __restrict__ x,
                                                     const float* __restrict__ wr,
                                                     int* __restrict__ tok_e,
                                                     float* __restrict__ tok_w,
                                                     int* __restrict__ counts) {
    const int t = blockIdx.x * 8 + (threadIdx.x >> 5);
    const int lane = threadIdx.x & 31;
    float lg[E_NUM] = {0.f,0.f,0.f,0.f,0.f,0.f,0.f,0.f};
    const float* xr = x + (size_t)t * C_DIM;
    for (int c = lane; c < C_DIM; c += 32) {
        const float xv = xr[c];
        const float* w = wr + c * E_NUM;
        #pragma unroll
        for (int e = 0; e < E_NUM; ++e) lg[e] += xv * w[e];
    }
    #pragma unroll
    for (int off = 16; off >= 1; off >>= 1) {
        #pragma unroll
        for (int e = 0; e < E_NUM; ++e) lg[e] += __shfl_xor(lg[e], off, 32);
    }
    // top-2 (jax tie-break: lowest index wins)
    int i0 = 0;
    #pragma unroll
    for (int e = 1; e < E_NUM; ++e) if (lg[e] > lg[i0]) i0 = e;
    int i1 = (i0 == 0) ? 1 : 0;
    #pragma unroll
    for (int e = 0; e < E_NUM; ++e) if (e != i0 && lg[e] > lg[i1]) i1 = e;
    // softmax normalization cancels in the top-k renorm: use exp(logit - max)
    const float p1 = __expf(lg[i1] - lg[i0]);   // p0 == 1
    const float s  = 1.0f + p1;
    if (lane == 0) {
        tok_e[2*t]   = i0; tok_w[2*t]   = 1.0f / s;
        tok_e[2*t+1] = i1; tok_w[2*t+1] = p1 / s;
        atomicAdd(&counts[i0], 1);
        atomicAdd(&counts[i1], 1);
    }
}

// ---------------- 4. scan + tile table (E=8, single thread) ----------------
__global__ void scan_tiles_kernel(const int* __restrict__ counts,
                                  int* __restrict__ offsets,
                                  int* __restrict__ tiles,
                                  int* __restrict__ ntiles) {
    int acc = 0, nt = 0;
    for (int e = 0; e < E_NUM; ++e) {
        offsets[e] = acc;
        const int c = counts[e];
        for (int j = 0; j < c; j += TILE_M) {
            tiles[2*nt]   = e;
            tiles[2*nt+1] = acc + j;
            ++nt;
        }
        acc += c;
    }
    offsets[E_NUM] = acc;
    *ntiles = nt;
}

// ---------------- 5. scatter tokens into per-expert segments ----------------
__global__ __launch_bounds__(256) void scatter_kernel(const int* __restrict__ tok_e,
                                                      const float* __restrict__ tok_w,
                                                      const int* __restrict__ offsets,
                                                      int* __restrict__ cursors,
                                                      int* __restrict__ assign_tok,
                                                      float* __restrict__ assign_w) {
    const int t = blockIdx.x * 256 + threadIdx.x;
    #pragma unroll
    for (int k = 0; k < 2; ++k) {
        const int e = tok_e[2*t + k];
        const int pos = atomicAdd(&cursors[e], 1);
        const int idx = offsets[e] + pos;
        assign_tok[idx] = t;
        assign_w[idx]   = tok_w[2*t + k];
    }
}

// ---------------- 6. fused expert FFN: 32-token tile, x + h in LDS (160KB) ----------
// LDS tiles use a 16B-chunk XOR swizzle: byte = row*rowBytes + ((chunk ^ (row&15))<<4)
// -> the 16 distinct A-fragment rows map to 16 distinct bank-quads (conflict-free b128).
__global__ __launch_bounds__(256) void expert_ffn_kernel(
        const unsigned short* __restrict__ xb,    // [N][C] bf16
        const unsigned short* __restrict__ w1t,   // [E][H][C] bf16 (B rows k-contig)
        const unsigned short* __restrict__ w2t,   // [E][C][H] bf16
        const float* __restrict__ b1,             // [E][H]
        const float* __restrict__ b2,             // [E][C]
        const int*  __restrict__ assign_tok,
        const float* __restrict__ assign_w,
        const int*  __restrict__ offsets,         // [E+1]
        const int*  __restrict__ tiles,           // [*][2]
        const int*  __restrict__ ntiles,
        float* __restrict__ out)                  // [N][C]
{
    if ((int)blockIdx.x >= *ntiles) return;

    __shared__ unsigned short hb[TILE_M * H_DIM];   // 128 KB: 32 x 2048 bf16, swizzled
    __shared__ unsigned short xs[TILE_M * C_DIM];   //  32 KB: 32 x  512 bf16, swizzled

    const int e         = tiles[2*blockIdx.x];
    const int row_start = tiles[2*blockIdx.x + 1];
    const int row_end   = offsets[e + 1];

    const int tid  = threadIdx.x;
    const int lane = tid & 31;
    const int wv   = tid >> 5;      // wave 0..7
    const int m    = lane & 15;     // fragment row / B column
    const int hi   = lane >> 4;     // half-wave select

    // ---- stage x tile (32 gathered rows) into LDS ----
    {
        const int r  = tid >> 3;                 // 0..31
        const int rr = row_start + r;
        const int tok = (rr < row_end) ? assign_tok[rr] : 0;
        const unsigned short* src = xb + (size_t)tok * C_DIM;
        const int c0 = tid & 7;
        #pragma unroll
        for (int j = 0; j < 8; ++j) {
            const int c = c0 + 8 * j;            // 16B chunk index, 0..63
            const v4u d = *(const v4u*)(src + c * 8);
            *(v4u*)((char*)xs + (r << 10) + (((c ^ (r & 15))) << 4)) = d;
        }
    }
    __syncthreads();

    const unsigned short* w1e = w1t + (size_t)e * H_DIM * C_DIM;
    const unsigned short* w2e = w2t + (size_t)e * C_DIM * H_DIM;

    // ---- GEMM1: h[32 x 2048] = relu(x @ W1 + b1); wave owns 256 H-cols ----
    for (int nt = 0; nt < 16; ++nt) {
        const int n0 = wv * 256 + nt * 16;
        const unsigned short* brow = w1e + (size_t)(n0 + m) * C_DIM;
        v8f acc0 = {0.f,0.f,0.f,0.f,0.f,0.f,0.f,0.f};
        v8f acc1 = {0.f,0.f,0.f,0.f,0.f,0.f,0.f,0.f};
        #pragma unroll 4
        for (int ks = 0; ks < 16; ++ks) {
            const int kb = ks * 32 + hi * 16;    // B: 16 contiguous K per lane
            const int ca = ks * 4  + hi * 2;     // A: first 16B chunk of 16 contiguous K
            Frag b, a0, a1;
            b.u[0]  = *(const v4u*)(brow + kb);
            b.u[1]  = *(const v4u*)(brow + kb + 8);
            a0.u[0] = *(const v4u*)((const char*)xs + (m << 10)        + (((ca    ) ^ m) << 4));
            a0.u[1] = *(const v4u*)((const char*)xs + (m << 10)        + (((ca + 1) ^ m) << 4));
            a1.u[0] = *(const v4u*)((const char*)xs + ((m + 16) << 10) + (((ca    ) ^ m) << 4));
            a1.u[1] = *(const v4u*)((const char*)xs + ((m + 16) << 10) + (((ca + 1) ^ m) << 4));
            acc0 = wmma_bf16(a0, b, acc0);
            acc1 = wmma_bf16(a1, b, acc1);
        }
        const float bias = b1[e * H_DIM + n0 + m];
        const int col = n0 + m;
        const int ch  = col >> 3;                // 16B chunk within h row
        const int within = (col & 7) * 2;
        #pragma unroll
        for (int v = 0; v < 8; ++v) {
            const int r0 = v + 8 * hi;           // msub0 row (0..15)
            float h0 = acc0[v] + bias; h0 = h0 > 0.f ? h0 : 0.f;
            float h1 = acc1[v] + bias; h1 = h1 > 0.f ? h1 : 0.f;
            *(unsigned short*)((char*)hb + (r0 << 12)        + ((ch ^ r0) << 4) + within) = f2bf(h0);
            *(unsigned short*)((char*)hb + ((r0 + 16) << 12) + ((ch ^ r0) << 4) + within) = f2bf(h1);
        }
    }
    __syncthreads();

    // ---- GEMM2: y[32 x 512] = h @ W2 + b2; wave owns 64 C-cols; K-outer, 8 acc tiles ----
    const unsigned short* brow2[4];
    #pragma unroll
    for (int nt = 0; nt < 4; ++nt)
        brow2[nt] = w2e + (size_t)(wv * 64 + nt * 16 + m) * H_DIM;

    v8f acc[8];
    #pragma unroll
    for (int i = 0; i < 8; ++i) acc[i] = (v8f){0.f,0.f,0.f,0.f,0.f,0.f,0.f,0.f};

    #pragma unroll 2
    for (int ks = 0; ks < 64; ++ks) {
        const int kb = ks * 32 + hi * 16;
        const int ca = ks * 4  + hi * 2;
        Frag a0, a1;
        a0.u[0] = *(const v4u*)((const char*)hb + (m << 12)        + (((ca    ) ^ m) << 4));
        a0.u[1] = *(const v4u*)((const char*)hb + (m << 12)        + (((ca + 1) ^ m) << 4));
        a1.u[0] = *(const v4u*)((const char*)hb + ((m + 16) << 12) + (((ca    ) ^ m) << 4));
        a1.u[1] = *(const v4u*)((const char*)hb + ((m + 16) << 12) + (((ca + 1) ^ m) << 4));
        #pragma unroll
        for (int nt = 0; nt < 4; ++nt) {
            Frag b;
            b.u[0] = *(const v4u*)(brow2[nt] + kb);
            b.u[1] = *(const v4u*)(brow2[nt] + kb + 8);
            acc[2*nt]     = wmma_bf16(a0, b, acc[2*nt]);
            acc[2*nt + 1] = wmma_bf16(a1, b, acc[2*nt + 1]);
        }
    }

    // ---- epilogue: gate-weight scale + atomic accumulate into out ----
    int   tokr[2][8];
    float wr_[2][8];
    #pragma unroll
    for (int msub = 0; msub < 2; ++msub) {
        #pragma unroll
        for (int v = 0; v < 8; ++v) {
            const int row = v + 8 * hi + 16 * msub;
            const int r = row_start + row;
            const bool ok = r < row_end;
            tokr[msub][v] = ok ? assign_tok[r] : 0;
            wr_[msub][v]  = ok ? assign_w[r]   : 0.0f;
        }
    }
    #pragma unroll
    for (int nt = 0; nt < 4; ++nt) {
        const int n0 = wv * 64 + nt * 16;
        const float bias = b2[e * C_DIM + n0 + m];
        #pragma unroll
        for (int msub = 0; msub < 2; ++msub) {
            #pragma unroll
            for (int v = 0; v < 8; ++v) {
                const float val = (acc[2*nt + msub][v] + bias) * wr_[msub][v];
                atomicAdd(&out[(size_t)tokr[msub][v] * C_DIM + n0 + m], val);
            }
        }
    }
}

// ---------------- host ----------------
extern "C" void kernel_launch(void* const* d_in, const int* in_sizes, int n_in,
                              void* d_out, int out_size, void* d_ws, size_t ws_size,
                              hipStream_t stream) {
    const float* x   = (const float*)d_in[0];
    const float* wr  = (const float*)d_in[1];
    const float* W1  = (const float*)d_in[2];
    const float* b1  = (const float*)d_in[3];
    const float* W2  = (const float*)d_in[4];
    const float* b2  = (const float*)d_in[5];
    float* out = (float*)d_out;

    // workspace layout (bytes)
    constexpr size_t SZ_XB  = (size_t)N_TOK * C_DIM * 2;              // 16 MB
    constexpr size_t SZ_W   = (size_t)E_NUM * C_DIM * H_DIM * 2;      // 16 MB each
    constexpr size_t O_XB   = 0;
    constexpr size_t O_W1T  = O_XB  + SZ_XB;
    constexpr size_t O_W2T  = O_W1T + SZ_W;
    constexpr size_t O_TOKE = O_W2T + SZ_W;                           // int[2N]
    constexpr size_t O_TOKW = O_TOKE + (size_t)2 * N_TOK * 4;
    constexpr size_t O_ATOK = O_TOKW + (size_t)2 * N_TOK * 4;
    constexpr size_t O_AW   = O_ATOK + (size_t)2 * N_TOK * 4;
    constexpr size_t O_CTRL = O_AW   + (size_t)2 * N_TOK * 4;
    // ctrl: counts[8] @+0, offsets[9] @+64, cursors[8] @+128, ntiles @+192
    constexpr size_t O_TILES = O_CTRL + 256;
    constexpr size_t TOTAL   = O_TILES + (size_t)2 * MAX_TILES * 4;
    if (ws_size < TOTAL) return;

    char* ws = (char*)d_ws;
    unsigned short* xb  = (unsigned short*)(ws + O_XB);
    unsigned short* w1t = (unsigned short*)(ws + O_W1T);
    unsigned short* w2t = (unsigned short*)(ws + O_W2T);
    int*   tok_e   = (int*)  (ws + O_TOKE);
    float* tok_w   = (float*)(ws + O_TOKW);
    int*   atok    = (int*)  (ws + O_ATOK);
    float* aw      = (float*)(ws + O_AW);
    int*   counts  = (int*)  (ws + O_CTRL);
    int*   offsets = (int*)  (ws + O_CTRL + 64);
    int*   cursors = (int*)  (ws + O_CTRL + 128);
    int*   ntiles  = (int*)  (ws + O_CTRL + 192);
    int*   tiles   = (int*)  (ws + O_TILES);

    hipMemsetAsync(out, 0, (size_t)out_size * sizeof(float), stream);
    hipMemsetAsync(ws + O_CTRL, 0, 256, stream);

    cvt_x_kernel<<<(N_TOK * C_DIM / 4) / 256, 256, 0, stream>>>(x, xb);
    transpose_bf16_kernel<<<dim3(H_DIM/32, C_DIM/32, E_NUM), 256, 0, stream>>>(W1, w1t, C_DIM, H_DIM);
    transpose_bf16_kernel<<<dim3(C_DIM/32, H_DIM/32, E_NUM), 256, 0, stream>>>(W2, w2t, H_DIM, C_DIM);
    router_kernel<<<N_TOK / 8, 256, 0, stream>>>(x, wr, tok_e, tok_w, counts);
    scan_tiles_kernel<<<1, 1, 0, stream>>>(counts, offsets, tiles, ntiles);
    scatter_kernel<<<N_TOK / 256, 256, 0, stream>>>(tok_e, tok_w, offsets, cursors, atok, aw);
    expert_ffn_kernel<<<MAX_TILES, 256, 0, stream>>>(xb, w1t, w2t, b1, b2,
                                                     atok, aw, offsets, tiles, ntiles, out);
}
